// Screen_36670430773889
// MI455X (gfx1250) — 8-bit matrix operand order repaired and checked by reading, compiled, not run on hardware
//
#include <hip/hip_runtime.h>
#include <stdint.h>

#define NUM_TILE  16
#define T         256
#define NPRIMS    65536
#define MAXP      16384
#define BLOCK_WH  68.0f
#define MAXVAL    15.99999f

typedef int v8i __attribute__((ext_vector_type(8)));

__device__ __forceinline__ void tileBounds(float x, float y, float r,
                                           int& xmn, int& xmx, int& ymn, int& ymx) {
  float fx = x / BLOCK_WH;
  float fy = y / BLOCK_WH;
  xmn = (int)fminf(fmaxf(fx - r, 0.0f), MAXVAL);
  xmx = (int)fminf(fmaxf(fx + r, 0.0f), MAXVAL);
  ymn = (int)fminf(fmaxf(fy - r, 0.0f), MAXVAL);
  ymx = (int)fminf(fmaxf(fy + r, 0.0f), MAXVAL);
}

__global__ void k_init(int* __restrict__ counts, int* __restrict__ cursors) {
  int i = threadIdx.x;
  if (i < T) { counts[i] = 0; cursors[i] = 0; }
}

#define WPB 4
__global__ void k_count_wmma(const float* __restrict__ pos2d,
                             const float* __restrict__ radius,
                             int* __restrict__ counts) {
  __shared__ unsigned short rowm[WPB][64];
  __shared__ unsigned short colm[WPB][64];
  const int lane   = threadIdx.x & 31;
  const int wave   = threadIdx.x >> 5;
  const int nWaves = gridDim.x * WPB;
  const int nChunks = NPRIMS >> 6;

  v8i acc = {0, 0, 0, 0, 0, 0, 0, 0};

  for (int base = blockIdx.x * WPB; base < nChunks; base += nWaves) {
    const int  chunk  = base + wave;
    const bool active = (chunk < nChunks);
    if (active) {
      int pn = ((chunk + nWaves) << 6) + lane;
      if (pn < NPRIMS) __builtin_prefetch(&pos2d[2 * pn], 0, 0);
      #pragma unroll
      for (int s = 0; s < 2; ++s) {
        int k = lane + 32 * s;
        int p = (chunk << 6) + k;
        float x = pos2d[2 * p], y = pos2d[2 * p + 1], r = radius[p];
        int xmn, xmx, ymn, ymx;
        tileBounds(x, y, r, xmn, xmx, ymn, ymx);
        colm[wave][k] = (unsigned short)(((1u << (xmx + 1)) - 1u) & ~((1u << xmn) - 1u));
        rowm[wave][k] = (unsigned short)(((1u << (ymx + 1)) - 1u) & ~((1u << ymn) - 1u));
      }
    }
    __syncthreads();
    if (active) {
      const int MN  = lane & 15;
      const int hiA = (lane >= 16) ? 8  : 0;
      const int hiB = (lane >= 16) ? 16 : 0;
      const int abase[8] = {0, 4, 16, 20, 32, 36, 48, 52};
      v8i A, B;
      #pragma unroll
      for (int v = 0; v < 8; ++v) {
        const int kA = abase[v] + hiA;
        const int kB = abase[v] + hiA;
        unsigned aw = 0, bw = 0;
        #pragma unroll
        for (int b = 0; b < 4; ++b) {
          aw |= (unsigned)((rowm[wave][kA + b] >> MN) & 1) << (8 * b);
          bw |= (unsigned)((colm[wave][kB + b] >> MN) & 1) << (8 * b);
        }
        A[v] = (int)aw;
        B[v] = (int)bw;
      }
      acc = __builtin_amdgcn_wmma_i32_16x16x64_iu8(false, A, false, B, acc,
                                                   false, false);
    }
    __syncthreads();
  }

  const int n     = lane & 15;
  const int mBase = (lane >= 16) ? 8 : 0;
  #pragma unroll
  for (int g = 0; g < 8; ++g) {
    int t = (g + mBase) * NUM_TILE + n;
    if (acc[g] != 0) atomicAdd(&counts[t], acc[g]);
  }
}

__global__ void k_scan(const int* __restrict__ counts,
                       int* __restrict__ out_counts,
                       int* __restrict__ offs) {
  if (threadIdx.x == 0) {
    int run = 0;
    for (int t = 0; t < T; ++t) {
      offs[t] = run;
      run += counts[t];
      out_counts[t] = run;
    }
  }
}

__global__ void k_fill(int* __restrict__ out_tail,
                       unsigned long long* __restrict__ entries) {
  long long idx    = (long long)blockIdx.x * blockDim.x + threadIdx.x;
  long long stride = (long long)gridDim.x * blockDim.x;
  const long long nOut = (long long)T * NPRIMS;
  for (long long i = idx; i < nOut; i += stride) out_tail[i] = -1;
  const long long nEnt = (long long)T * MAXP;
  for (long long i = idx; i < nEnt; i += stride) entries[i] = ~0ull;
}

__global__ void k_scatter(const float* __restrict__ pos2d,
                          const float* __restrict__ radius,
                          const float* __restrict__ depth,
                          int* __restrict__ cursors,
                          unsigned long long* __restrict__ entries) {
  int p = blockIdx.x * blockDim.x + threadIdx.x;
  if (p >= NPRIMS) return;
  float x = pos2d[2 * p], y = pos2d[2 * p + 1], r = radius[p];
  int xmn, xmx, ymn, ymx;
  tileBounds(x, y, r, xmn, xmx, ymn, ymx);
  unsigned db = __float_as_uint(depth[p]);
  unsigned long long key = ((unsigned long long)db << 32) | (unsigned)p;
  for (int yi = ymn; yi <= ymx; ++yi) {
    for (int xi = xmn; xi <= xmx; ++xi) {
      int t = yi * NUM_TILE + xi;
      int j = atomicAdd(&cursors[t], 1);
      if (j < MAXP) entries[(long long)t * MAXP + j] = key;
    }
  }
}

__global__ void k_sort(const int* __restrict__ counts,
                       const int* __restrict__ offs,
                       const unsigned long long* __restrict__ entries,
                       int* __restrict__ out_tail) {
  extern __shared__ unsigned long long s[];
  const int t = blockIdx.x;
  int c = counts[t];
  if (c > MAXP) c = MAXP;
  if (c <= 0) return;
  const int off = offs[t];
  const unsigned long long* seg = entries + (long long)t * MAXP;

  int m = 1;
  while (m < c) m <<= 1;
  if (m < 2) {
    if (threadIdx.x == 0) out_tail[off] = (int)(seg[0] & 0xFFFFFFFFull);
    return;
  }
  for (int i = threadIdx.x; i < m; i += blockDim.x) s[i] = seg[i];
  __syncthreads();

  for (int k = 2; k <= m; k <<= 1) {
    for (int j = k >> 1; j > 0; j >>= 1) {
      for (int i = threadIdx.x; i < m; i += blockDim.x) {
        int ixj = i ^ j;
        if (ixj > i) {
          unsigned long long a = s[i], b = s[ixj];
          bool up = ((i & k) == 0);
          if ((a > b) == up) { s[i] = b; s[ixj] = a; }
        }
      }
      __syncthreads();
    }
  }
  for (int i = threadIdx.x; i < c; i += blockDim.x)
    out_tail[off + i] = (int)(s[i] & 0xFFFFFFFFull);
}

extern "C" void kernel_launch(void* const* d_in, const int* in_sizes, int n_in,
                              void* d_out, int out_size, void* d_ws, size_t ws_size,
                              hipStream_t stream) {
  (void)in_sizes; (void)n_in; (void)out_size; (void)ws_size;
  const float* pos2d  = (const float*)d_in[0];
  const float* radius = (const float*)d_in[1];
  const float* depth  = (const float*)d_in[2];
  int* out = (int*)d_out;

  char* ws = (char*)d_ws;
  int* counts  = (int*)(ws);
  int* offs    = (int*)(ws + 1024);
  int* cursors = (int*)(ws + 2048);
  unsigned long long* entries = (unsigned long long*)(ws + 4096);

  k_init<<<1, 256, 0, stream>>>(counts, cursors);
  k_count_wmma<<<32, 128, 0, stream>>>(pos2d, radius, counts);
  k_scan<<<1, 32, 0, stream>>>(counts, out, offs);
  k_fill<<<2048, 256, 0, stream>>>(out + T, entries);
  k_scatter<<<NPRIMS / 256, 256, 0, stream>>>(pos2d, radius, depth, cursors, entries);
  k_sort<<<T, 256, (size_t)MAXP * sizeof(unsigned long long), stream>>>(counts, offs, entries, out + T);
}
